// RecurrentAttention_49890340110536
// MI455X (gfx1250) — matrix operand order repaired and checked by reading, not compiled, not run on hardware
//
#include <hip/hip_runtime.h>
#include <hip/hip_bf16.h>
#include <math.h>
#include <stdint.h>

// Sizes fixed by the reference
#define BATCH 2
#define SEQ   2048
#define DMODEL 1024
#define NHEAD 16
#define HDIM  64
#define ROWS  (BATCH * SEQ)   // 4096

typedef _Float16 half8  __attribute__((ext_vector_type(8)));
typedef _Float16 v16h   __attribute__((ext_vector_type(16)));
typedef float    v8f    __attribute__((ext_vector_type(8)));

__device__ __forceinline__ v16h make_v16(half8 lo, half8 hi) {
    return __builtin_shufflevector(lo, hi, 0,1,2,3,4,5,6,7,8,9,10,11,12,13,14,15);
}

__device__ __forceinline__ v8f wmma_f16(v16h a, v16h b, v8f c) {
    // (neg_a, A, neg_b, B, c_mod, C, reuse_a, reuse_b)
    return __builtin_amdgcn_wmma_f32_16x16x32_f16(false, a, false, b, (short)0, c, false, false);
}

// Async global->LDS copy of 16 bytes per lane (GLOBAL_LOAD_ASYNC_TO_LDS_B128,
// tracked by ASYNCcnt). IOFFSET is added to BOTH the LDS and global address
// (ISA 08_async_tensor 4.4), so offset:16 copies the second half8 of a 32B row.
__device__ __forceinline__ void async_lds_b128(const _Float16* gsrc, _Float16* ldst) {
    uint64_t ga = (uint64_t)(uintptr_t)gsrc;
    uint32_t la = (uint32_t)(uintptr_t)ldst;   // generic-ptr low 32 bits == LDS offset
    asm volatile("global_load_async_to_lds_b128 %0, %1, off" :: "v"(la), "v"(ga) : "memory");
}
__device__ __forceinline__ void async_lds_b128x2(const _Float16* gsrc, _Float16* ldst) {
    uint64_t ga = (uint64_t)(uintptr_t)gsrc;
    uint32_t la = (uint32_t)(uintptr_t)ldst;
    asm volatile("global_load_async_to_lds_b128 %0, %1, off"           :: "v"(la), "v"(ga) : "memory");
    asm volatile("global_load_async_to_lds_b128 %0, %1, off offset:16" :: "v"(la), "v"(ga) : "memory");
}
__device__ __forceinline__ void wait_async0() {
    asm volatile("s_wait_asynccnt 0x0" ::: "memory");
}
__device__ __forceinline__ void wait_ds0() {
    asm volatile("s_wait_dscnt 0x0" ::: "memory");
}

// ---------------------------------------------------------------------------
// Kernel 1: projection  out[B,H,S,hd](f16) = (X[4096,1024] @ W[1024,1024] + b) * scale
// 256 threads (8 waves), 128x64 output tile, register-side software pipeline
// (fp32->f16 conversion forces the VGPR staging path).
// ---------------------------------------------------------------------------
__global__ __launch_bounds__(256) void proj_kernel(
    const float* __restrict__ X, const float* __restrict__ W,
    const float* __restrict__ bias, _Float16* __restrict__ out, float scale)
{
    __shared__ __align__(16) _Float16 As[128 * 40];  // A tile 128x32, stride 40
    __shared__ __align__(16) _Float16 Bt[64 * 40];   // B tile transposed [n][k]

    const int tid  = threadIdx.x;
    const int lane = tid & 31;
    const int w    = tid >> 5;
    const int ln   = lane & 15;
    const int hs   = lane >> 4;       // half-wave select
    const int row0 = blockIdx.x * 128;
    const int col0 = blockIdx.y * 64;

    const int mA  = tid >> 1, koA = (tid & 1) * 16;       // A staging coords
    const int nB  = tid & 63, kbB = (tid >> 6) * 8;       // B staging coords

    v8f acc[4] = {};
    half8 pa0, pa1, pb;

    auto loadTiles = [&](int k0) {
        const float* srcA = X + (size_t)(row0 + mA) * DMODEL + k0 + koA;
        #pragma unroll
        for (int i = 0; i < 8; ++i) { pa0[i] = (_Float16)srcA[i]; pa1[i] = (_Float16)srcA[8 + i]; }
        #pragma unroll
        for (int i = 0; i < 8; ++i)
            pb[i] = (_Float16)W[(size_t)(k0 + kbB + i) * DMODEL + col0 + nB];
    };

    loadTiles(0);
    for (int k0 = 0; k0 < DMODEL; k0 += 32) {
        __syncthreads();
        *(half8*)&As[mA * 40 + koA]     = pa0;
        *(half8*)&As[mA * 40 + koA + 8] = pa1;
        *(half8*)&Bt[nB * 40 + kbB]     = pb;
        __syncthreads();
        if (k0 + 32 < DMODEL) loadTiles(k0 + 32);   // overlap next global fetch with WMMAs

        const int arow = w * 16 + ln;
        v16h aF = make_v16(*(const half8*)&As[arow * 40 + hs * 8],
                           *(const half8*)&As[arow * 40 + hs * 8 + 16]);
        v16h bF[4];
        #pragma unroll
        for (int ct = 0; ct < 4; ++ct) {
            const int n = ct * 16 + ln;
            bF[ct] = make_v16(*(const half8*)&Bt[n * 40 + hs * 8],
                              *(const half8*)&Bt[n * 40 + hs * 8 + 16]);
        }
        #pragma unroll
        for (int ct = 0; ct < 4; ++ct) acc[ct] = wmma_f16(aF, bF[ct], acc[ct]);
    }

    #pragma unroll
    for (int ct = 0; ct < 4; ++ct) {
        const int n  = col0 + ct * 16 + ln;
        const float bv = bias[n];
        const int h = n >> 6, d = n & 63;
        #pragma unroll
        for (int r = 0; r < 8; ++r) {
            const int m = row0 + w * 16 + r + hs * 8;    // C layout: VGPR r -> row r / r+8
            const int b = m >> 11, s = m & (SEQ - 1);
            const float v = (acc[ct][r] + bv) * scale;
            out[(((size_t)(b * NHEAD + h)) * SEQ + s) * HDIM + d] = (_Float16)v;
        }
    }
}

// ---------------------------------------------------------------------------
// Kernel 2: block-causal flash attention.
// 256 threads (8 waves) per (b, h, 128 query rows); wave owns 16 q rows.
// K tile: double-buffered LDS fed by global_load_async_to_lds_b128, issued one
// iteration ahead.  V tile: register prefetch + manual transpose into LDS.
// ---------------------------------------------------------------------------
__global__ __launch_bounds__(256) void attn_kernel(
    const _Float16* __restrict__ qw, const _Float16* __restrict__ kw,
    const _Float16* __restrict__ vw, _Float16* __restrict__ ow)
{
    __shared__ __align__(16) _Float16 Kt[2][64 * 72];    // [key][d]  (B-frag for Q.K^T)
    __shared__ __align__(16) _Float16 Vt[2][64 * 72];    // [d][key]  (B-frag for P.V)
    __shared__ __align__(16) _Float16 Pb[8 * 16 * 72];   // per-wave P tile

    const int tid  = threadIdx.x;
    const int lane = tid & 31;
    const int w    = tid >> 5;
    const int ln   = lane & 15;
    const int hs   = lane >> 4;

    const int qb = blockIdx.x & 15;
    const int h  = (blockIdx.x >> 4) & 15;
    const int b  = blockIdx.x >> 8;
    const size_t headBase = ((size_t)(b * NHEAD + h)) * SEQ * HDIM;
    const int q0   = qb * 128;
    const int kend = ((q0 >> 8) + 1) << 8;   // block-causal visible extent
    const int nIter = kend >> 6;

    // Q fragments (1/sqrt(hd) already folded into Q)
    const _Float16* qrow = qw + headBase + (size_t)(q0 + w * 16 + ln) * HDIM;
    v16h qF0 = make_v16(*(const half8*)&qrow[hs * 8],      *(const half8*)&qrow[hs * 8 + 16]);
    v16h qF1 = make_v16(*(const half8*)&qrow[32 + hs * 8], *(const half8*)&qrow[32 + hs * 8 + 16]);

    v8f o[4] = {};
    float mstat[8], lstat[8];
    #pragma unroll
    for (int r = 0; r < 8; ++r) { mstat[r] = -3.0e38f; lstat[r] = 0.0f; }

    _Float16* pw = &Pb[w * 16 * 72];

    const int keyS  = tid >> 2;
    const int doffS = (tid & 3) * 16;
    const _Float16* kbase = kw + headBase + (size_t)keyS * HDIM + doffS;
    const _Float16* vbase = vw + headBase + (size_t)keyS * HDIM + doffS;

    half8 vA, vB;
    auto loadV = [&](int kb) {
        const _Float16* vs = vbase + (size_t)kb * HDIM;
        vA = *(const half8*)vs;
        vB = *(const half8*)(vs + 8);
    };

    async_lds_b128x2(kbase, &Kt[0][keyS * 72 + doffS]);   // K tile 0 in flight
    loadV(0);

    for (int it = 0; it < nIter; ++it) {
        const int buf = it & 1;
        _Float16* Ktb = Kt[buf];
        _Float16* Vtb = Vt[buf];

        // Transpose-stage V for this iteration (Vt[buf] last read two iterations ago)
        #pragma unroll
        for (int i = 0; i < 8; ++i) {
            Vtb[(doffS + i) * 72 + keyS]     = vA[i];
            Vtb[(doffS + 8 + i) * 72 + keyS] = vB[i];
        }
        wait_async0();        // my K-tile async copies for this buffer landed
        __syncthreads();      // everyone's K async + V stores visible

        if (it + 1 < nIter) { // next tile in flight while we do the WMMAs
            const int kb = (it + 1) * 64;
            async_lds_b128x2(kbase + (size_t)kb * HDIM, &Kt[buf ^ 1][keyS * 72 + doffS]);
            loadV(kb);
        }

        // Scores: S[16 q x 64 keys]; batch fragment loads, then WMMAs
        v8f s[4];
        #pragma unroll
        for (int kp = 0; kp < 2; ++kp) {
            const int k0t = (2 * kp) * 16 + ln;
            const int k1t = (2 * kp + 1) * 16 + ln;
            v16h b00 = make_v16(*(const half8*)&Ktb[k0t * 72 + hs * 8],
                                *(const half8*)&Ktb[k0t * 72 + hs * 8 + 16]);
            v16h b01 = make_v16(*(const half8*)&Ktb[k0t * 72 + 32 + hs * 8],
                                *(const half8*)&Ktb[k0t * 72 + 32 + hs * 8 + 16]);
            v16h b10 = make_v16(*(const half8*)&Ktb[k1t * 72 + hs * 8],
                                *(const half8*)&Ktb[k1t * 72 + hs * 8 + 16]);
            v16h b11 = make_v16(*(const half8*)&Ktb[k1t * 72 + 32 + hs * 8],
                                *(const half8*)&Ktb[k1t * 72 + 32 + hs * 8 + 16]);
            v8f z0 = {}, z1 = {};
            z0 = wmma_f16(qF0, b00, z0);
            z1 = wmma_f16(qF0, b10, z1);
            s[2 * kp]     = wmma_f16(qF1, b01, z0);
            s[2 * kp + 1] = wmma_f16(qF1, b11, z1);
        }

        // Online softmax; row r lives across 16 lanes of one half-wave
        // (xor masks 1,2,4,8 never cross bit 4).
        #pragma unroll
        for (int r = 0; r < 8; ++r) {
            float rm = fmaxf(fmaxf(s[0][r], s[1][r]), fmaxf(s[2][r], s[3][r]));
            rm = fmaxf(rm, __shfl_xor(rm, 1));
            rm = fmaxf(rm, __shfl_xor(rm, 2));
            rm = fmaxf(rm, __shfl_xor(rm, 4));
            rm = fmaxf(rm, __shfl_xor(rm, 8));
            const float mnew = fmaxf(mstat[r], rm);
            const float corr = __expf(mstat[r] - mnew);
            mstat[r] = mnew;
            float rs = 0.0f;
            #pragma unroll
            for (int kt = 0; kt < 4; ++kt) {
                const float p = __expf(s[kt][r] - mnew);
                s[kt][r] = p;
                rs += p;
            }
            rs += __shfl_xor(rs, 1);
            rs += __shfl_xor(rs, 2);
            rs += __shfl_xor(rs, 4);
            rs += __shfl_xor(rs, 8);
            lstat[r] = lstat[r] * corr + rs;
            #pragma unroll
            for (int ct = 0; ct < 4; ++ct) o[ct][r] *= corr;
        }

        // C-layout P -> A-fragment layout via per-wave LDS round trip
        #pragma unroll
        for (int kt = 0; kt < 4; ++kt) {
            const int key = kt * 16 + ln;
            #pragma unroll
            for (int r = 0; r < 8; ++r)
                pw[(r + hs * 8) * 72 + key] = (_Float16)s[kt][r];
        }
        wait_ds0();   // LDS in-order per wave; this blocks compiler reordering too

        const _Float16* pr = &pw[ln * 72];
        v16h pF0 = make_v16(*(const half8*)&pr[hs * 8],      *(const half8*)&pr[hs * 8 + 16]);
        v16h pF1 = make_v16(*(const half8*)&pr[32 + hs * 8], *(const half8*)&pr[32 + hs * 8 + 16]);

        // O += P x V  (batch V fragments per pair, then WMMAs)
        #pragma unroll
        for (int cp = 0; cp < 2; ++cp) {
            const int d0 = (2 * cp) * 16 + ln;
            const int d1 = (2 * cp + 1) * 16 + ln;
            v16h v00 = make_v16(*(const half8*)&Vtb[d0 * 72 + hs * 8],
                                *(const half8*)&Vtb[d0 * 72 + hs * 8 + 16]);
            v16h v01 = make_v16(*(const half8*)&Vtb[d0 * 72 + 32 + hs * 8],
                                *(const half8*)&Vtb[d0 * 72 + 32 + hs * 8 + 16]);
            v16h v10 = make_v16(*(const half8*)&Vtb[d1 * 72 + hs * 8],
                                *(const half8*)&Vtb[d1 * 72 + hs * 8 + 16]);
            v16h v11 = make_v16(*(const half8*)&Vtb[d1 * 72 + 32 + hs * 8],
                                *(const half8*)&Vtb[d1 * 72 + 32 + hs * 8 + 16]);
            o[2 * cp]     = wmma_f16(pF1, v01, wmma_f16(pF0, v00, o[2 * cp]));
            o[2 * cp + 1] = wmma_f16(pF1, v11, wmma_f16(pF0, v10, o[2 * cp + 1]));
        }
    }

    // Finalize: divide by l, store f16 in [B,S,H,hd] (== row-major [4096,1024])
    #pragma unroll
    for (int ct = 0; ct < 4; ++ct) {
        const int d = ct * 16 + ln;
        #pragma unroll
        for (int r = 0; r < 8; ++r) {
            const int srow = q0 + w * 16 + r + hs * 8;
            const float val = o[ct][r] / lstat[r];
            ow[((size_t)(b * SEQ + srow) * NHEAD + h) * HDIM + d] = (_Float16)val;
        }
    }
}

// ---------------------------------------------------------------------------
// Kernel 3: output projection  out[4096,1024](f32) = Ain[4096,1024](f16) @ Wo + bo
// A tile staged with async-to-LDS (already f16); W via register prefetch.
// ---------------------------------------------------------------------------
__global__ __launch_bounds__(256) void outproj_kernel(
    const _Float16* __restrict__ Ain, const float* __restrict__ W,
    const float* __restrict__ bias, float* __restrict__ out)
{
    __shared__ __align__(16) _Float16 As[128 * 40];
    __shared__ __align__(16) _Float16 Bt[64 * 40];

    const int tid  = threadIdx.x;
    const int lane = tid & 31;
    const int w    = tid >> 5;
    const int ln   = lane & 15;
    const int hs   = lane >> 4;
    const int row0 = blockIdx.x * 128;
    const int col0 = blockIdx.y * 64;

    const int mA  = tid >> 1, koA = (tid & 1) * 16;
    const int nB  = tid & 63, kbB = (tid >> 6) * 8;
    const _Float16* abase = Ain + (size_t)(row0 + mA) * DMODEL + koA;

    v8f acc[4] = {};
    half8 pb;

    auto loadB = [&](int k0) {
        #pragma unroll
        for (int i = 0; i < 8; ++i)
            pb[i] = (_Float16)W[(size_t)(k0 + kbB + i) * DMODEL + col0 + nB];
    };

    loadB(0);
    for (int k0 = 0; k0 < DMODEL; k0 += 32) {
        __syncthreads();                                  // prior reads of As/Bt done
        async_lds_b128x2(abase + k0, &As[mA * 40 + koA]); // A: global->LDS, no VGPR trip
        *(half8*)&Bt[nB * 40 + kbB] = pb;
        wait_async0();
        __syncthreads();
        if (k0 + 32 < DMODEL) loadB(k0 + 32);

        const int arow = w * 16 + ln;
        v16h aF = make_v16(*(const half8*)&As[arow * 40 + hs * 8],
                           *(const half8*)&As[arow * 40 + hs * 8 + 16]);
        v16h bF[4];
        #pragma unroll
        for (int ct = 0; ct < 4; ++ct) {
            const int n = ct * 16 + ln;
            bF[ct] = make_v16(*(const half8*)&Bt[n * 40 + hs * 8],
                              *(const half8*)&Bt[n * 40 + hs * 8 + 16]);
        }
        #pragma unroll
        for (int ct = 0; ct < 4; ++ct) acc[ct] = wmma_f16(aF, bF[ct], acc[ct]);
    }

    #pragma unroll
    for (int ct = 0; ct < 4; ++ct) {
        const int n  = col0 + ct * 16 + ln;
        const float bv = bias[n];
        #pragma unroll
        for (int r = 0; r < 8; ++r) {
            const int m = row0 + w * 16 + r + hs * 8;
            out[(size_t)m * DMODEL + n] = acc[ct][r] + bv;
        }
    }
}

// ---------------------------------------------------------------------------
extern "C" void kernel_launch(void* const* d_in, const int* in_sizes, int n_in,
                              void* d_out, int out_size, void* d_ws, size_t ws_size,
                              hipStream_t stream) {
    const float* query = (const float*)d_in[0];
    const float* key   = (const float*)d_in[1];
    const float* value = (const float*)d_in[2];
    const float* Wq = (const float*)d_in[3];
    const float* bq = (const float*)d_in[4];
    const float* Wk = (const float*)d_in[5];
    const float* bk = (const float*)d_in[6];
    const float* Wv = (const float*)d_in[7];
    const float* bv = (const float*)d_in[8];
    const float* Wo = (const float*)d_in[9];
    const float* bo = (const float*)d_in[10];

    // Workspace (f16): Q | K | V | attn-out, each B*H*S*hd = 4,194,304 elems (~33.5 MB total)
    const size_t HEADTOT = (size_t)BATCH * NHEAD * SEQ * HDIM;
    _Float16* ws = (_Float16*)d_ws;
    _Float16* qw = ws;
    _Float16* kw = ws + HEADTOT;
    _Float16* vw = ws + 2 * HEADTOT;
    _Float16* aw = ws + 3 * HEADTOT;

    const dim3 gProj(ROWS / 128, DMODEL / 64);   // 32 x 16
    const float qscale = 0.125f;                 // 1/sqrt(64), folded into Q

    proj_kernel<<<gProj, 256, 0, stream>>>(query, Wq, bq, qw, qscale);
    proj_kernel<<<gProj, 256, 0, stream>>>(key,   Wk, bk, kw, 1.0f);
    proj_kernel<<<gProj, 256, 0, stream>>>(value, Wv, bv, vw, 1.0f);

    attn_kernel<<<BATCH * NHEAD * (SEQ / 128), 256, 0, stream>>>(qw, kw, vw, aw);

    outproj_kernel<<<gProj, 256, 0, stream>>>(aw, Wo, bo, (float*)d_out);
}
